// step_shGLM_16097537425996
// MI455X (gfx1250) — matrix operand order repaired and checked by reading, not compiled, not run on hardware
//
#include <hip/hip_runtime.h>
#include <math.h>

typedef __attribute__((ext_vector_type(16))) _Float16 v16h;
typedef __attribute__((ext_vector_type(8)))  float    v8f;

#define SUB   20
#define NSS   19     // sub_no - 1
#define TSYN  200
#define THIST 50
#define TILE  256

// ---------------------------------------------------------------------------
// Kernel 1: build small constant tables (alpha kernels, spike-prop matrix)
// ---------------------------------------------------------------------------
__global__ void shglm_build_consts(const float* __restrict__ C_den,
                                   const float* __restrict__ W_ns_syn,
                                   const float* __restrict__ Delta,
                                   const float* __restrict__ W_ns_hist,
                                   const float* __restrict__ W_s_prop,
                                   const float* __restrict__ W_ns_prop,
                                   float* __restrict__ ke, float* __restrict__ ki,
                                   float* __restrict__ hk, float* __restrict__ pk,
                                   float* __restrict__ A) {
    const float tau0 = 1.0f, tau1 = expf(1.0f), tau2 = expf(2.0f);
    // synaptic kernels: syn_kern[s,c,t] = sum_b W_ns_syn[s,b,c]*u*exp(-u),
    // u = max(t - exp(Delta[s,c]), 0)/tau_b
    for (int idx = threadIdx.x; idx < SUB * 2 * TSYN; idx += blockDim.x) {
        int t = idx % TSYN; int sc = idx / TSYN; int c = sc & 1; int s = sc >> 1;
        float delta = expf(Delta[s * 2 + c]);
        float ts = fmaxf((float)t - delta, 0.0f);
        float u0 = ts / tau0, u1 = ts / tau1, u2 = ts / tau2;
        float v = W_ns_syn[s * 6 + 0 * 2 + c] * u0 * expf(-u0)
                + W_ns_syn[s * 6 + 1 * 2 + c] * u1 * expf(-u1)
                + W_ns_syn[s * 6 + 2 * 2 + c] * u2 * expf(-u2);
        if (c == 0) ke[s * TSYN + t] = v; else ki[s * TSYN + t] = v;
    }
    // history kernel hk[c,j] (19 x 50)
    for (int idx = threadIdx.x; idx < NSS * THIST; idx += blockDim.x) {
        int j = idx % THIST; int c = idx / THIST;
        float u0 = j / tau0, u1 = j / tau1, u2 = j / tau2;
        hk[idx] = W_ns_hist[c * 3 + 0] * u0 * expf(-u0)
                + W_ns_hist[c * 3 + 1] * u1 * expf(-u1)
                + W_ns_hist[c * 3 + 2] * u2 * expf(-u2);
    }
    // propagation kernel pk[s,j] (20 x 50)
    for (int idx = threadIdx.x; idx < SUB * THIST; idx += blockDim.x) {
        int j = idx % THIST; int s = idx / THIST;
        float u0 = j / tau0, u1 = j / tau1, u2 = j / tau2;
        pk[idx] = W_ns_prop[s * 3 + 0] * u0 * expf(-u0)
                + W_ns_prop[s * 3 + 1] * u1 * expf(-u1)
                + W_ns_prop[s * 3 + 2] * u2 * expf(-u2);
    }
    // A[s,c] = C_s[s,c] * W_s_prop[c],  C_s = C_den[1:,1:]
    for (int idx = threadIdx.x; idx < NSS * NSS; idx += blockDim.x) {
        int c = idx % NSS; int s = idx / NSS;
        A[idx] = C_den[(1 + s) * SUB + (1 + c)] * W_s_prop[c];
    }
}

// ---------------------------------------------------------------------------
// Kernel 2: in = S(TxK) @ C^T(Kx20) via v_wmma_f32_16x16x32_f16.
// One wave computes a 16-row x 32-col tile (cols 0..19 stored). Requires
// T % 16 == 0 (grid = T/16). Main K-loop is unguarded (vectorizable b128
// loads); single guarded tail iteration handles K % 32.
// Lanes with col1 >= 20 accumulate garbage in acc1 that is never stored.
// ---------------------------------------------------------------------------
__global__ void shglm_proj_wmma(const float* __restrict__ S,
                                const float* __restrict__ C,   // (20, K) row-major
                                float* __restrict__ out,       // (T, 20)
                                int K) {
    const int m0   = blockIdx.x * 16;
    const int lane = threadIdx.x;            // 0..31
    const int hi   = lane >> 4;
    const int rowA = m0 + (lane & 15);
    const int kA   = hi << 4;                // A frag K-base: 0 or 16
    const int nB   = lane & 15;              // B frag column
    const int kB   = hi << 4;                // B frag K-base: 0 or 16
    const int n1   = 16 + nB;

    const float* Arow  = S + (size_t)rowA * K;
    const float* B0row = C + (size_t)nB * K;
    const float* B1row = C + (size_t)(n1 < SUB ? n1 : 0) * K;  // clamp: junk cols unstored

    v8f acc0 = {}; v8f acc1 = {};
    const int Kmain = K & ~31;
    int k0 = 0;
    for (; k0 < Kmain; k0 += 32) {
        v16h a, b0, b1;
        const float* Ap = Arow + k0 + kA;
        const float* Bp0 = B0row + k0 + kB;
        const float* Bp1 = B1row + k0 + kB;
#pragma unroll
        for (int i = 0; i < 8; ++i) a[i]     = (_Float16)Ap[i];
#pragma unroll
        for (int i = 0; i < 8; ++i) a[8 + i] = (_Float16)Ap[8 + i];
#pragma unroll
        for (int e = 0; e < 16; ++e) { b0[e] = (_Float16)Bp0[e]; b1[e] = (_Float16)Bp1[e]; }
        acc0 = __builtin_amdgcn_wmma_f32_16x16x32_f16(false, a, false, b0,
                                                      (short)0, acc0, false, false);
        acc1 = __builtin_amdgcn_wmma_f32_16x16x32_f16(false, a, false, b1,
                                                      (short)0, acc1, false, false);
    }
    if (k0 < K) {   // tail: K % 32 elements, guarded
        v16h a, b0, b1;
#pragma unroll
        for (int i = 0; i < 8; ++i) {
            int ka = k0 + kA + i;
            a[i]     = (ka < K)      ? (_Float16)Arow[ka]      : (_Float16)0.0f;
            a[8 + i] = (ka + 8 < K) ? (_Float16)Arow[ka + 8] : (_Float16)0.0f;
        }
#pragma unroll
        for (int e = 0; e < 16; ++e) {
            int kb = k0 + kB + e;
            b0[e] = (kb < K) ? (_Float16)B0row[kb] : (_Float16)0.0f;
            b1[e] = (kb < K) ? (_Float16)B1row[kb] : (_Float16)0.0f;
        }
        acc0 = __builtin_amdgcn_wmma_f32_16x16x32_f16(false, a, false, b0,
                                                      (short)0, acc0, false, false);
        acc1 = __builtin_amdgcn_wmma_f32_16x16x32_f16(false, a, false, b1,
                                                      (short)0, acc1, false, false);
    }
    // C/D layout: VGPR r, lanes 0-15 -> M=r, N=lane; lanes 16-31 -> M=8+r
#pragma unroll
    for (int r = 0; r < 8; ++r) {
        int row  = m0 + r + (hi << 3);
        int col0 = lane & 15;
        out[(size_t)row * SUB + col0] = acc0[r];
        int col1 = 16 + col0;
        if (col1 < SUB) out[(size_t)row * SUB + col1] = acc1[r];
    }
}

// ---------------------------------------------------------------------------
// Kernel 3: depthwise causal conv (K=200) for syn_ns + elementwise syn_s
// ---------------------------------------------------------------------------
__global__ void shglm_synconv(const float* __restrict__ in_e,
                              const float* __restrict__ in_i,
                              const float* __restrict__ ke,
                              const float* __restrict__ ki,
                              const float* __restrict__ W_s_syn,
                              float* __restrict__ syn_ns,
                              float* __restrict__ syn_s, int T) {
    const int s  = blockIdx.y;
    const int t0 = blockIdx.x * TILE;
    __shared__ float se[TILE + TSYN - 1], si[TILE + TSYN - 1];
    __shared__ float k_e[TSYN], k_i[TSYN];
    for (int i = threadIdx.x; i < TSYN; i += blockDim.x) {
        k_e[i] = ke[s * TSYN + i];
        k_i[i] = ki[s * TSYN + i];
    }
    for (int i = threadIdx.x; i < TILE + TSYN - 1; i += blockDim.x) {
        int tau = t0 - (TSYN - 1) + i;
        bool ok = (tau >= 0) && (tau < T);
        se[i] = ok ? in_e[(size_t)tau * SUB + s] : 0.0f;
        si[i] = ok ? in_i[(size_t)tau * SUB + s] : 0.0f;
    }
    __syncthreads();
    const int t = t0 + threadIdx.x;
    if (t < T) {
        const int base = threadIdx.x + (TSYN - 1);   // LDS index of tau == t
        float acc = 0.0f;
        for (int k = 0; k < TSYN; ++k)
            acc += se[base - k] * k_e[k] + si[base - k] * k_i[k];
        syn_ns[(size_t)t * SUB + s] = acc;
        syn_s[(size_t)t * SUB + s] = se[base] * W_s_syn[s * 2 + 0]
                                   + si[base] * W_s_syn[s * 2 + 1];
    }
}

// ---------------------------------------------------------------------------
// Kernel 4: sequential spike scan. One wave; lane = spiking subunit (0..18).
// Spikes packed in a ballot bitmask; recurrent matvec = 19 masked adds.
// Next step's synaptic drive is prefetched before this step's ALU work.
// ---------------------------------------------------------------------------
__global__ void shglm_spike_scan(const float* __restrict__ syn_s,
                                 const float* __restrict__ A,
                                 const float* __restrict__ Theta_s,
                                 const float* __restrict__ decay,
                                 float* __restrict__ Zout,
                                 float* __restrict__ Xout, int T) {
    const int lane = threadIdx.x;
    const bool act = lane < NSS;
    float a[NSS];
#pragma unroll
    for (int c = 0; c < NSS; ++c) a[c] = act ? A[lane * NSS + c] : 0.0f;
    const float th  = act ? Theta_s[lane] : 0.0f;
    const float dec = act ? decay[lane]   : 0.0f;
    float x = 0.0f;
    unsigned mask = 0u;   // spikes from previous step

    auto body = [&](int t, float syn) {
        float prop = 0.0f;
#pragma unroll
        for (int c = 0; c < NSS; ++c)
            prop += (mask & (1u << c)) ? a[c] : 0.0f;
        float xin = x * dec + syn + prop + th;
        bool z = act && (xin >= 0.0f);
        mask = __builtin_amdgcn_ballot_w32(z);
        x = z ? 0.0f : xin;
        if (act) {
            Zout[(size_t)t * NSS + lane] = z ? 1.0f : 0.0f;
            Xout[(size_t)t * NSS + lane] = x;
        }
    };

    float syn = act ? syn_s[1 + lane] : 0.0f;
    for (int t = 0; t + 1 < T; ++t) {
        float syn_next = act ? syn_s[(size_t)(t + 1) * SUB + 1 + lane] : 0.0f;
        body(t, syn);
        syn = syn_next;
    }
    body(T - 1, syn);
}

// ---------------------------------------------------------------------------
// Kernel 5: pre[t,s] = syn_ns + f_prop + Theta_ns + (s>0)*f_hist, fully
// parallel given Z. f_prop[t,s] = sum_{d=1..50} ZC[t-d,s]*pk[s,d-1] with
// ZC[tau,s] = sum_c Z[tau,c]*C_den[s,1+c]; f_hist analogous on Z[:,s-1].
// ---------------------------------------------------------------------------
__global__ void shglm_pre_ns(const float* __restrict__ Z,
                             const float* __restrict__ C_den,
                             const float* __restrict__ pk,
                             const float* __restrict__ hk,
                             const float* __restrict__ syn_ns,
                             const float* __restrict__ Theta_ns,
                             float* __restrict__ pre, int T) {
    const int s  = blockIdx.y;
    const int t0 = blockIdx.x * TILE;
    __shared__ float zc[TILE + THIST], zh[TILE + THIST];
    __shared__ float pks[THIST], hks[THIST];
    float cd[NSS];
#pragma unroll
    for (int c = 0; c < NSS; ++c) cd[c] = C_den[s * SUB + 1 + c];
    for (int i = threadIdx.x; i < THIST; i += blockDim.x) {
        pks[i] = pk[s * THIST + i];
        hks[i] = (s >= 1) ? hk[(s - 1) * THIST + i] : 0.0f;
    }
    for (int i = threadIdx.x; i < TILE + THIST; i += blockDim.x) {
        int tau = t0 - THIST + i;
        float acc = 0.0f, zv = 0.0f;
        if (tau >= 0 && tau < T) {
#pragma unroll
            for (int c = 0; c < NSS; ++c) acc += Z[(size_t)tau * NSS + c] * cd[c];
            if (s >= 1) zv = Z[(size_t)tau * NSS + (s - 1)];
        }
        zc[i] = acc; zh[i] = zv;
    }
    __syncthreads();
    const int t = t0 + threadIdx.x;
    if (t < T) {
        const int base = threadIdx.x + THIST;   // LDS index of tau == t
        float fp = 0.0f, fh = 0.0f;
#pragma unroll
        for (int d = 1; d <= THIST; ++d) {
            fp += zc[base - d] * pks[d - 1];
            fh += zh[base - d] * hks[d - 1];
        }
        pre[(size_t)t * SUB + s] = syn_ns[(size_t)t * SUB + s] + fp + Theta_ns[s] + fh;
    }
}

// ---------------------------------------------------------------------------
// Kernel 6: sequential Y scan. One wave; lane = subunit (0..19). Y_prev is
// broadcast via __shfl at constant lane indices (v_readlane) -- no LDS, no
// barriers. pre[] is prefetched one step ahead.
// ---------------------------------------------------------------------------
__global__ void shglm_y_scan(const float* __restrict__ pre,
                             const float* __restrict__ C_den,
                             const float* __restrict__ W_ns_sub,
                             const float* __restrict__ V_o,
                             float* __restrict__ fv,
                             float* __restrict__ Yns, int T) {
    const int lane = threadIdx.x;
    const bool act = lane < SUB;
    float cd[SUB];
#pragma unroll
    for (int c = 0; c < SUB; ++c) cd[c] = act ? C_den[lane * SUB + c] : 0.0f;
    const float wsub = act ? W_ns_sub[lane] : 0.0f;
    const float vo = V_o[0];
    float y = 0.0f;

    auto body = [&](int t, float p) {
        float raw = 0.0f;
#pragma unroll
        for (int c = 0; c < SUB; ++c) raw += cd[c] * __shfl(y, c, 32);
        float xin = p + raw;
        y = act ? (wsub / (1.0f + __expf(-xin))) : 0.0f;
        if (act) {
            if (lane == 0) fv[t] = y + vo;
            else           Yns[(size_t)t * NSS + (lane - 1)] = y;
        }
    };

    float p = act ? pre[lane] : 0.0f;
    for (int t = 0; t + 1 < T; ++t) {
        float p_next = act ? pre[(size_t)(t + 1) * SUB + lane] : 0.0f;
        body(t, p);
        p = p_next;
    }
    body(T - 1, p);
}

// ---------------------------------------------------------------------------
extern "C" void kernel_launch(void* const* d_in, const int* in_sizes, int n_in,
                              void* d_out, int out_size, void* d_ws, size_t ws_size,
                              hipStream_t stream) {
    const float* S_e      = (const float*)d_in[0];
    const float* S_i      = (const float*)d_in[1];
    const float* C_den    = (const float*)d_in[2];
    const float* C_syn_e  = (const float*)d_in[3];
    const float* C_syn_i  = (const float*)d_in[4];
    const float* W_s_syn  = (const float*)d_in[5];
    const float* W_ns_syn = (const float*)d_in[6];
    const float* Delta    = (const float*)d_in[7];
    const float* W_ns_sub = (const float*)d_in[8];
    const float* V_o      = (const float*)d_in[9];
    const float* Theta_s  = (const float*)d_in[10];
    const float* Theta_ns = (const float*)d_in[11];
    const float* W_ns_hist= (const float*)d_in[12];
    const float* W_s_prop = (const float*)d_in[13];
    const float* W_ns_prop= (const float*)d_in[14];
    const float* decay    = (const float*)d_in[15];

    const int E = in_sizes[3] / SUB;          // 2000
    const int I = in_sizes[4] / SUB;          // 500
    const int T = in_sizes[0] / E;            // 10000 (multiple of 16)

    float* ws   = (float*)d_ws;
    float* ke   = ws;                         // 20*200
    float* ki   = ws + 4000;                  // 20*200
    float* hk   = ws + 8000;                  // 19*50
    float* pk   = ws + 8960;                  // 20*50
    float* A    = ws + 9984;                  // 19*19
    float* in_e = ws + 10368;                 // T*20
    float* in_i   = in_e   + (size_t)T * SUB;
    float* syn_ns = in_i   + (size_t)T * SUB;
    float* syn_s  = syn_ns + (size_t)T * SUB;
    float* pre    = syn_s  + (size_t)T * SUB;

    float* out = (float*)d_out;
    float* fv  = out;                         // (T,)
    float* Yns = out + T;                     // (T,19)
    float* Z   = Yns + (size_t)T * NSS;       // (T,19)
    float* X   = Z   + (size_t)T * NSS;       // (T,19)

    shglm_build_consts<<<1, 256, 0, stream>>>(C_den, W_ns_syn, Delta, W_ns_hist,
                                              W_s_prop, W_ns_prop, ke, ki, hk, pk, A);
    shglm_proj_wmma<<<T / 16, 32, 0, stream>>>(S_e, C_syn_e, in_e, E);
    shglm_proj_wmma<<<T / 16, 32, 0, stream>>>(S_i, C_syn_i, in_i, I);
    shglm_synconv<<<dim3((T + TILE - 1) / TILE, SUB), TILE, 0, stream>>>(
        in_e, in_i, ke, ki, W_s_syn, syn_ns, syn_s, T);
    shglm_spike_scan<<<1, 32, 0, stream>>>(syn_s, A, Theta_s, decay, Z, X, T);
    shglm_pre_ns<<<dim3((T + TILE - 1) / TILE, SUB), TILE, 0, stream>>>(
        Z, C_den, pk, hk, syn_ns, Theta_ns, pre, T);
    shglm_y_scan<<<1, 32, 0, stream>>>(pre, C_den, W_ns_sub, V_o, fv, Yns, T);
}
